// SentenceSelfAttention_69638599737532
// MI455X (gfx1250) — matrix operand order repaired and checked by reading, not compiled, not run on hardware
//
#include <hip/hip_runtime.h>

typedef _Float16 f16;
typedef __attribute__((ext_vector_type(16))) _Float16 v16h;
typedef __attribute__((ext_vector_type(8)))  _Float16 v8h;
typedef __attribute__((ext_vector_type(4)))  _Float16 v4h;
typedef __attribute__((ext_vector_type(8)))  float    v8f;

#define SS 256   // sequence length
#define EE 128   // embedding dim
#define NSENT 512

struct __align__(16) Smem {
    f16   x [SS][EE];        // 64 KB  row-major sentence (logits-B source)
    f16   xT[EE][SS];        // 64 KB  transposed sentence (PV-B source)
    f16   Wt[EE][EE];        // 32 KB  Wt[f][e] = W[e][f]  (xw-B source)
    f16   xw[SS][EE];        // 64 KB  projected queries (staged for A-fragment reload)
    f16   P [8][16][32];     //  8 KB  per-wave prob chunk (C->A layout conversion)
    float ostage[8][16][EE]; // 64 KB  per-wave output tile (coalesced store staging)
};                           // total 296 KB of the 320 KB WGP LDS

// fast tanh: 1 - 2/(e^{2x}+1) via v_exp_f32 + v_rcp_f32 (feeds a softmax; ~1ulp is ample)
__device__ __forceinline__ float fast_tanh(float x) {
    return 1.0f - 2.0f * __builtin_amdgcn_rcpf(__expf(2.0f * x) + 1.0f);
}

__global__ __launch_bounds__(256)
void sent_attn_wmma(const float* __restrict__ x_g,
                    const float* __restrict__ W_g,
                    const int*   __restrict__ vlen_g,
                    float*       __restrict__ out_g)
{
    __shared__ Smem sm;

    const int sent = blockIdx.x;          // one workgroup per (b,d) sentence
    const int tid  = threadIdx.x;         // 256 threads = 8 wave32
    const int lane = tid & 31;
    const int wave = tid >> 5;
    const int half = lane >> 4;           // which 16-lane half
    const int l16  = lane & 15;

    const int    len = vlen_g[sent];
    const float* xg  = x_g + (size_t)sent * (SS * EE);

    // ---------- Phase 0: global -> LDS (f32 -> f16), wide b128 loads ----------
    {
        const float4* xg4 = (const float4*)xg;
        for (int i4 = tid; i4 < SS * EE / 4; i4 += 256) {
            int s  = i4 >> 5;              // 32 float4 per row
            int e4 = (i4 & 31) * 4;
            float4 v = xg4[i4];
            v4h h = { (f16)v.x, (f16)v.y, (f16)v.z, (f16)v.w };
            *(v4h*)&sm.x[s][e4] = h;       // packed b64 LDS write
            sm.xT[e4 + 0][s] = h[0];
            sm.xT[e4 + 1][s] = h[1];
            sm.xT[e4 + 2][s] = h[2];
            sm.xT[e4 + 3][s] = h[3];
        }
        const float4* Wg4 = (const float4*)W_g;
        for (int i4 = tid; i4 < EE * EE / 4; i4 += 256) {
            int e  = i4 >> 5;
            int f4 = (i4 & 31) * 4;
            float4 v = Wg4[i4];
            sm.Wt[f4 + 0][e] = (f16)v.x;
            sm.Wt[f4 + 1][e] = (f16)v.y;
            sm.Wt[f4 + 2][e] = (f16)v.z;
            sm.Wt[f4 + 3][e] = (f16)v.w;
        }
    }
    __syncthreads();

    // A-fragment (16x32 f16): half=0 -> K 0..15 ; half=1 -> K 16..31 (contiguous)
    auto loadA_E = [&](const f16 (*M)[EE], int rbase, int c) -> v16h {
        int row = rbase + l16;
        int k0  = c * 32 + half * 16;
        v8h lo = *(const v8h*)&M[row][k0];
        v8h hi = *(const v8h*)&M[row][k0 + 8];
        v16h a;
#pragma unroll
        for (int i = 0; i < 8; ++i) { a[i] = lo[i]; a[i + 8] = hi[i]; }
        return a;
    };
    // B-fragment (32x16 f16): lane N = l16, K = half*16 .. half*16+15 (contiguous)
    auto loadB_E = [&](const f16 (*M)[EE], int nbase, int kbase) -> v16h {
        return *(const v16h*)&M[nbase + l16][kbase + half * 16];
    };
    auto loadB_S = [&](const f16 (*M)[SS], int nbase, int kbase) -> v16h {
        return *(const v16h*)&M[nbase + l16][kbase + half * 16];
    };

    // ---------- Phase 1: xw = x @ W  (each wave: its own 2 query tiles) ----------
    for (int q = 0; q < 2; ++q) {
        const int rbase = (wave + q * 8) * 16;
        v16h ax[4];
#pragma unroll
        for (int c = 0; c < 4; ++c) ax[c] = loadA_E(sm.x, rbase, c);
#pragma unroll
        for (int n = 0; n < 8; ++n) {
            v8f acc = {};
#pragma unroll
            for (int c = 0; c < 4; ++c) {
                v16h b = loadB_E(sm.Wt, n * 16, c * 32);
                acc = __builtin_amdgcn_wmma_f32_16x16x32_f16(
                          false, ax[c], false, b, (short)0, acc, false, false);
            }
            // C layout: row = v + 8*half, col = l16 ; stage to LDS as f16
#pragma unroll
            for (int v = 0; v < 8; ++v)
                sm.xw[rbase + v + half * 8][n * 16 + l16] = (f16)acc[v];
        }
    }
    // xw rows for a query tile are produced and consumed by the same wave: no barrier.

    // ---------- Phase 2: fused tanh-softmax-attention per query tile ----------
    for (int q = 0; q < 2; ++q) {
        const int rbase = (wave + q * 8) * 16;
        v16h axw[4];
#pragma unroll
        for (int c = 0; c < 4; ++c) axw[c] = loadA_E(sm.xw, rbase, c);

        v8f  O[8] = {};                 // 16x128 f32 output accumulator
        float rowsum[8];
#pragma unroll
        for (int v = 0; v < 8; ++v) rowsum[v] = 0.0f;

        for (int kc = 0; kc < 8; ++kc) {            // key chunks of 32
#pragma unroll
            for (int h = 0; h < 2; ++h) {           // two 16-key logits tiles
                const int kt = kc * 2 + h;
                v8f lg = {};
#pragma unroll
                for (int c = 0; c < 4; ++c) {
                    v16h b = loadB_E(sm.x, kt * 16, c * 32);
                    lg = __builtin_amdgcn_wmma_f32_16x16x32_f16(
                             false, axw[c], false, b, (short)0, lg, false, false);
                }
                const int  key = kt * 16 + l16;     // mask depends only on key column
                const bool ok  = key < len;
#pragma unroll
                for (int v = 0; v < 8; ++v) {
                    float t  = fast_tanh(lg[v]);    // |t|<=1 -> exp needs no max-sub
                    float ev = ok ? __expf(t) : 0.0f;
                    rowsum[v] += ev;
                    sm.P[wave][v + half * 8][h * 16 + l16] = (f16)ev;
                }
            }
            // reload P chunk as A-fragment (layout conversion via LDS, same wave)
            v16h ap;
            {
                int k0 = half * 16;
                v8h lo = *(const v8h*)&sm.P[wave][l16][k0];
                v8h hi = *(const v8h*)&sm.P[wave][l16][k0 + 8];
#pragma unroll
                for (int i = 0; i < 8; ++i) { ap[i] = lo[i]; ap[i + 8] = hi[i]; }
            }
#pragma unroll
            for (int n = 0; n < 8; ++n) {           // O += P @ x[kc*32 .. +31, :]
                v16h b = loadB_S(sm.xT, n * 16, kc * 32);
                O[n] = __builtin_amdgcn_wmma_f32_16x16x32_f16(
                           false, ap, false, b, (short)0, O[n], false, false);
            }
        }

        // reduce row sums across the 16 lanes holding each row (xor stays in-half)
#pragma unroll
        for (int v = 0; v < 8; ++v) {
            float s = rowsum[v];
            s += __shfl_xor(s, 1, 32);
            s += __shfl_xor(s, 2, 32);
            s += __shfl_xor(s, 4, 32);
            s += __shfl_xor(s, 8, 32);
            rowsum[v] = 1.0f / (s + 1e-8f);
        }

        // normalize into per-wave LDS staging tile (same-wave, no barrier needed)
#pragma unroll
        for (int n = 0; n < 8; ++n)
#pragma unroll
            for (int v = 0; v < 8; ++v)
                sm.ostage[wave][v + half * 8][n * 16 + l16] = O[n][v] * rowsum[v];

        // coalesced b128 stores: wave's 16x128 tile is contiguous in the output
        float*        og = out_g + (size_t)sent * (SS * EE) + (size_t)rbase * EE;
        const float4* st = (const float4*)&sm.ostage[wave][0][0];
        float4*       og4 = (float4*)og;
#pragma unroll
        for (int i = 0; i < 16; ++i)
            og4[i * 32 + lane] = st[i * 32 + lane];
    }
}

extern "C" void kernel_launch(void* const* d_in, const int* in_sizes, int n_in,
                              void* d_out, int out_size, void* d_ws, size_t ws_size,
                              hipStream_t stream) {
    const float* words_enc = (const float*)d_in[0];
    const float* W         = (const float*)d_in[1];
    const int*   vlen      = (const int*)d_in[2];
    float*       out       = (float*)d_out;
    (void)in_sizes; (void)n_in; (void)out_size; (void)d_ws; (void)ws_size;

    sent_attn_wmma<<<NSENT, 256, 0, stream>>>(words_enc, W, vlen, out);
}
